// gruaging_rnnode_70308614636166
// MI455X (gfx1250) — matrix operand order repaired and checked by reading, not compiled, not run on hardware
//
#include <hip/hip_runtime.h>
#include <hip/hip_bf16.h>
#include <stddef.h>
#include <stdint.h>

// ---------------- problem constants ----------------
#define NI      16
#define NIP     17          // NINPUT + 1
#define NH      256         // NHIDDEN == NHAGE
#define NG      768         // 3 * NHAGE
#define TSTEP   501
#define BATCH   256
#define FPIT    100
#define NOUT    8
#define NDP     69888       // NI*NH + NH*NH + NH

// ---------------- WMMA vector types ----------------
typedef __attribute__((ext_vector_type(16))) __bf16          bf16x16;
typedef __attribute__((ext_vector_type(16))) unsigned short  u16x16;
typedef __attribute__((ext_vector_type(8)))  float           f32x8;

union FragU { u16x16 u; bf16x16 b; };

__device__ __forceinline__ unsigned short f2bf(float f) {
    unsigned u = __float_as_uint(f);
    u += 0x7FFFu + ((u >> 16) & 1u);        // round-to-nearest-even
    return (unsigned short)(u >> 16);
}

__device__ __forceinline__ bf16x16 load_frag(const unsigned short* p) {
    FragU f;
    f.u = *reinterpret_cast<const u16x16*>(p);   // 32B contiguous per lane
    return f.b;
}

__device__ __forceinline__ f32x8 wmma_bf16(bf16x16 a, bf16x16 b, f32x8 c) {
    return __builtin_amdgcn_wmma_f32_16x16x32_bf16(
        /*neg_a=*/false, a, /*neg_b=*/false, b,
        /*c_mod=*/(short)0, c, /*reuse_a=*/false, /*reuse_b=*/false);
}

// Async global->LDS path (CDNA5 GLOBAL_LOAD_ASYNC_TO_LDS_*, ASYNCcnt).
// Builtin signature (from clang diagnostic): takes pointers to
// 'int __attribute__((vector_size(16)))' in AS(1) (global) / AS(3) (LDS).
#if __has_builtin(__builtin_amdgcn_global_load_async_to_lds_b128) && \
    __has_builtin(__builtin_amdgcn_s_wait_asynccnt)
#define HAVE_ASYNC_LDS 1
typedef int v4i_vs __attribute__((vector_size(16)));
typedef __attribute__((address_space(1))) v4i_vs* gptr_v4i;
typedef __attribute__((address_space(3))) v4i_vs* lptr_v4i;
#define AS1P(p) ((gptr_v4i)(uintptr_t)(p))
#define AS3P(p) ((lptr_v4i)(uint32_t)(uintptr_t)(p))
#else
#define HAVE_ASYNC_LDS 0
#endif

// A-operand (16x32 bf16) element mapping, ISA 7.12.2:
//   lane l: m = l&15 ; e<8 -> k = (l>>4)*8 + e ; e>=8 -> k = 16 + (l>>4)*8 + (e-8)
// Inverse for k-in-chunk kl (0..31): lane_hi=(kl>>3)&1, e=(kl&7)+((kl>>4)<<3)

// ============================================================
// Kernel 1: pack GRU weights f32 -> bf16 B-operand fragments
//   B-frag layout: lane l holds column n = tile*16+(l&15),
//   16 consecutive K starting at kc*32 + (l>>4)*16.
// ============================================================
__global__ void pack_weights_k(const float* __restrict__ gWhh,   // (768,256)
                               const float* __restrict__ gWih,   // (768,17)
                               unsigned short* __restrict__ whh_f,
                               unsigned short* __restrict__ wih_f) {
    int idx = blockIdx.x * blockDim.x + threadIdx.x;
    const int NWHH = 48 * 8 * 512;     // 196608
    const int NWIH = 48 * 512;         // 24576
    if (idx < NWHH) {
        int frag = idx >> 9, r = idx & 511;
        int lane = r >> 4, j = r & 15;
        int ntile = frag >> 3, kc = frag & 7;
        int n = ntile * 16 + (lane & 15);
        int k = kc * 32 + ((lane >> 4) << 4) + j;
        whh_f[idx] = f2bf(gWhh[n * NH + k]);
    } else if (idx < NWHH + NWIH) {
        int i2 = idx - NWHH;
        int frag = i2 >> 9, r = i2 & 511;
        int lane = r >> 4, j = r & 15;
        int n = frag * 16 + (lane & 15);
        int k = ((lane >> 4) << 4) + j;               // K padded 17 -> 32
        wih_f[i2] = (k < NIP) ? f2bf(gWih[n * NIP + k]) : (unsigned short)0;
    }
}

// ============================================================
// Kernel 2: GRU forward. 16 WGs x 256 thr; WG = 16-batch tile,
// loops all 501 steps with h kept in LDS (f32 + bf16 A-frags).
// ============================================================
__global__ void gru_k(const float* __restrict__ u,          // (B,T,17)
                      const unsigned short* __restrict__ whh_f,
                      const unsigned short* __restrict__ wih_f,
                      const float* __restrict__ gbih,       // (768)
                      const float* __restrict__ gbhh,       // (768)
                      unsigned short* __restrict__ ht_frags) { // 16*8*512
    extern __shared__ float smem[];
    float* h_s  = smem;                        // 16*256
    float* gi_s = smem + 4096;                 // 16*768
    float* gh_s = smem + 16384;                // 16*768
    unsigned short* hfrag = (unsigned short*)(smem + 28672);   // 8*512, 32B aligned
    unsigned short* xfrag = hfrag + 4096;                      // 512

    const int tid  = threadIdx.x;
    const int w    = tid >> 5;
    const int lane = tid & 31;
    const int bt   = blockIdx.x;
    const int b0   = bt * 16;

    for (int i = tid; i < 16 * NH; i += 256) h_s[i] = 0.f;
    for (int i = tid; i < 8 * 512; i += 256) hfrag[i] = 0;

    const int j = tid;   // hidden column owned in update phase
    const float bir = gbih[j],        bhr = gbhh[j];
    const float biz = gbih[NH + j],   bhz = gbhh[NH + j];
    const float bin = gbih[2*NH + j], bhn = gbhh[2*NH + j];
    __syncthreads();

    for (int t = 0; t < TSTEP; ++t) {
        // ---- build input A-fragment (16x32, K padded from 17) ----
        #pragma unroll
        for (int q = 0; q < 2; ++q) {
            int idx = tid * 2 + q;
            int l2 = idx >> 4, e = idx & 15;
            int m = l2 & 15;
            int k = ((l2 >> 4) << 4) + e;
            float v = (k < NIP) ? u[(size_t)(b0 + m) * TSTEP * NIP + (size_t)t * NIP + k] : 0.f;
            xfrag[idx] = f2bf(v);
        }
        __syncthreads();

        // ---- WMMA phase: gi = x@Wih^T, gh = h@Whh^T ----
        #pragma unroll 1
        for (int s = 0; s < 6; ++s) {
            int ntile = w * 6 + s;
            f32x8 acch = {0.f,0.f,0.f,0.f,0.f,0.f,0.f,0.f};
            #pragma unroll
            for (int kc = 0; kc < 8; ++kc) {
                bf16x16 a  = load_frag(&hfrag[kc * 512 + lane * 16]);
                bf16x16 bb = load_frag(&whh_f[((size_t)(ntile * 8 + kc)) * 512 + lane * 16]);
                acch = wmma_bf16(a, bb, acch);
            }
            f32x8 acci = {0.f,0.f,0.f,0.f,0.f,0.f,0.f,0.f};
            {
                bf16x16 ax = load_frag(&xfrag[lane * 16]);
                bf16x16 bx = load_frag(&wih_f[(size_t)ntile * 512 + lane * 16]);
                acci = wmma_bf16(ax, bx, acci);
            }
            int col = ntile * 16 + (lane & 15);
            int mh  = (lane >> 4) * 8;
            #pragma unroll
            for (int r = 0; r < 8; ++r) {
                gh_s[(mh + r) * NG + col] = acch[r];
                gi_s[(mh + r) * NG + col] = acci[r];
            }
        }
        __syncthreads();

        // ---- gate nonlinearity + state update (thread j owns column j) ----
        #pragma unroll 4
        for (int m = 0; m < 16; ++m) {
            float ir  = gi_s[m * NG + j]          + bir;
            float hr  = gh_s[m * NG + j]          + bhr;
            float iz  = gi_s[m * NG + NH + j]     + biz;
            float hz  = gh_s[m * NG + NH + j]     + bhz;
            float inn = gi_s[m * NG + 2*NH + j]   + bin;
            float hn  = gh_s[m * NG + 2*NH + j]   + bhn;
            float rr = 1.f / (1.f + __expf(-(ir + hr)));
            float zz = 1.f / (1.f + __expf(-(iz + hz)));
            float nn = tanhf(inn + rr * hn);
            float hnew = (1.f - zz) * nn + zz * h_s[m * NH + j];
            h_s[m * NH + j] = hnew;
            // write back into A-fragment order for next step's WMMA
            int kc = j >> 5, kl = j & 31;
            int lhi = kl >> 4;
            int e   = kl & 15;
            hfrag[kc * 512 + (m + (lhi << 4)) * 16 + e] = f2bf(hnew);
        }
        // ordering to next iter's WMMA reads is covered by the post-xfrag barrier
    }
    __syncthreads();
    for (int i = tid; i < 4096; i += 256)
        ht_frags[(size_t)bt * 4096 + i] = hfrag[i];
}

// ============================================================
// Kernel 3: dparams GEMM (256 x 69888, K=256) via WMMA,
// fused epilogue builds A = Whh+Dwr, Bm = Wih+Dwu, C = b+Dbu.
// grid = (546 col-strips, 16 batch-tiles), 8 waves = 8 N-tiles.
// ============================================================
__global__ void hyper_k(const unsigned short* __restrict__ ht_frags,
                        const float* __restrict__ Wage,   // (69888,256)
                        const float* __restrict__ bage,
                        const float* __restrict__ Wih,    // (256,16)
                        const float* __restrict__ Whh,    // (256,256)
                        const float* __restrict__ bih,
                        const float* __restrict__ bhh,
                        float* __restrict__ A_ws,         // (B,65536)
                        float* __restrict__ Bm_ws,        // (B,4096)
                        float* __restrict__ C_ws) {       // (B,256)
    const int tid = threadIdx.x;
    const int w = tid >> 5, lane = tid & 31;
    const int bt = blockIdx.y;
    const int ntile = blockIdx.x * 8 + w;
    const int ncol0 = ntile * 16;

    f32x8 acc = {0.f,0.f,0.f,0.f,0.f,0.f,0.f,0.f};
    const int n = ncol0 + (lane & 15);
    #pragma unroll 1
    for (int kc = 0; kc < 8; ++kc) {
        bf16x16 a = load_frag(&ht_frags[((size_t)(bt * 8 + kc)) * 512 + lane * 16]);
        int kb = kc * 32 + ((lane >> 4) << 4);
        const float* wr = &Wage[(size_t)n * NH + kb];
        __builtin_prefetch(wr + NH, 0, 1);            // global_prefetch_b8
        FragU f;
        #pragma unroll
        for (int jj = 0; jj < 16; ++jj) f.u[jj] = f2bf(wr[jj]);
        acc = wmma_bf16(a, f.b, acc);
    }

    const float bg = bage[n];
    #pragma unroll
    for (int r = 0; r < 8; ++r) {
        int b = bt * 16 + (lane >> 4) * 8 + r;
        float dp = acc[r] + bg;
        if (n < NI * NH) {                                   // Dwu -> Bm
            Bm_ws[(size_t)b * (NI * NH) + n] = dp + Wih[n];
        } else if (n < NI * NH + NH * NH) {                  // Dwr -> A
            int c = n - NI * NH;
            A_ws[(size_t)b * (NH * NH) + c] = dp + Whh[c];
        } else {                                             // Dbu -> C
            int c = n - (NI * NH + NH * NH);
            C_ws[(size_t)b * NH + c] = dp + bih[c] + bhh[c];
        }
    }
}

// ============================================================
// Kernel 4: fixed point + Euler + output, one WG per batch.
// A (256KB f32) resident in LDS (320KB/WGP), staged via the
// CDNA5 async global->LDS engine (ASYNCcnt) when available.
// Rotated float4 LDS mat-vec avoids stride-256 bank conflicts.
// ============================================================
__device__ __forceinline__ float matvec_row(const float* A_s, const float* x_s, int h) {
    float y = 0.f;
    int k0 = (h * 4) & 255;                    // per-thread rotation
    #pragma unroll 8
    for (int kk = 0; kk < 64; ++kk) {
        int k = (k0 + kk * 4) & 255;
        const float4 a4 = *reinterpret_cast<const float4*>(A_s + h * NH + k);
        const float4 x4 = *reinterpret_cast<const float4*>(x_s + k);
        y += a4.x * x4.x + a4.y * x4.y + a4.z * x4.z + a4.w * x4.w;
    }
    return y;
}

__device__ __forceinline__ void write_out_row(const float* __restrict__ Wout,
                                              const float* __restrict__ bout,
                                              const float* x_s, float* __restrict__ out,
                                              int b, int row) {
    int w = threadIdx.x >> 5, lane = threadIdx.x & 31;   // 8 waves = 8 outputs
    float p = 0.f;
    #pragma unroll
    for (int q = 0; q < 8; ++q) {
        int hh = lane + 32 * q;
        p += Wout[w * NH + hh] * x_s[hh];
    }
    #pragma unroll
    for (int off = 16; off > 0; off >>= 1) p += __shfl_down(p, off, 32);
    if (lane == 0) out[((size_t)b * TSTEP + row) * NOUT + w] = p + bout[w];
}

__global__ void dyn_k(const float* __restrict__ u,
                      const float* __restrict__ A_ws,
                      const float* __restrict__ Bm_ws,
                      const float* __restrict__ C_ws,
                      const float* __restrict__ tau,
                      const float* __restrict__ Wout,
                      const float* __restrict__ bout,
                      float* __restrict__ out) {
    extern __shared__ float smem[];
    float* A_s  = smem;                 // 65536
    float* Bm_s = smem + 65536;         // 4096
    float* x_s  = smem + 69632;         // 256
    float* u_t  = smem + 69888;         // 16

    const int b = blockIdx.x;
    const int h = threadIdx.x;

    // ---- stage A (256KB) + Bm (16KB) into LDS ----
#if HAVE_ASYNC_LDS
    {
        // Per-lane async DMA: 16B per lane per op, no VGPR data movement;
        // overlaps with the scalar prologue below until s_wait_asynccnt 0.
        const char* ga = (const char*)(A_ws + (size_t)b * (NH * NH));
        char*       la = (char*)A_s;
        #pragma unroll 4
        for (int i = h; i < (NH * NH) / 4; i += 256)
            __builtin_amdgcn_global_load_async_to_lds_b128(
                AS1P(ga + (size_t)i * 16), AS3P(la + (size_t)i * 16), 0, 0);
        const char* gb = (const char*)(Bm_ws + (size_t)b * (NI * NH));
        char*       lb = (char*)Bm_s;
        #pragma unroll
        for (int i = h; i < (NI * NH) / 4; i += 256)
            __builtin_amdgcn_global_load_async_to_lds_b128(
                AS1P(gb + (size_t)i * 16), AS3P(lb + (size_t)i * 16), 0, 0);
    }
#else
    {
        const float4* src = reinterpret_cast<const float4*>(A_ws + (size_t)b * (NH * NH));
        float4* dst = reinterpret_cast<float4*>(A_s);
        for (int i = h; i < (NH * NH) / 4; i += 256) dst[i] = src[i];
        const float4* s2 = reinterpret_cast<const float4*>(Bm_ws + (size_t)b * (NI * NH));
        float4* d2 = reinterpret_cast<float4*>(Bm_s);
        for (int i = h; i < (NI * NH) / 4; i += 256) d2[i] = s2[i];
    }
#endif

    const float Cv  = C_ws[(size_t)b * NH + h];
    const float etv = __expf(tau[h]);
    const float ietv = 1.f / etv;
    x_s[h] = 0.f;
    if (h < NI) u_t[h] = u[(size_t)b * TSTEP * NIP + h];    // u[b,0,:16]

#if HAVE_ASYNC_LDS
    __builtin_amdgcn_s_wait_asynccnt(0);
#endif
    __syncthreads();

    float bu0 = 0.f;
    #pragma unroll
    for (int i = 0; i < NI; ++i) bu0 += Bm_s[h * NI + i] * u_t[i];

    // ---- fixed point: x = e^tau * tanh(A x + Bu0 + C) ----
    for (int it = 0; it < FPIT; ++it) {
        float y = matvec_row(A_s, x_s, h);
        __syncthreads();
        x_s[h] = etv * tanhf(y + bu0 + Cv);
        __syncthreads();
    }
    write_out_row(Wout, bout, x_s, out, b, 0);

    // ---- Euler: x += tanh(A x + Bu_t + C) - x/e^tau ----
    for (int t = 0; t < TSTEP - 1; ++t) {
        if (h < NI) u_t[h] = u[(size_t)b * TSTEP * NIP + (size_t)t * NIP + h];
        __syncthreads();
        float bu = 0.f;
        #pragma unroll
        for (int i = 0; i < NI; ++i) bu += Bm_s[h * NI + i] * u_t[i];
        float y  = matvec_row(A_s, x_s, h);
        float xo = x_s[h];
        __syncthreads();
        x_s[h] = xo + tanhf(y + bu + Cv) - xo * ietv;
        __syncthreads();
        write_out_row(Wout, bout, x_s, out, b, t + 1);
    }
}

// ============================================================
extern "C" void kernel_launch(void* const* d_in, const int* in_sizes, int n_in,
                              void* d_out, int out_size, void* d_ws, size_t ws_size,
                              hipStream_t stream) {
    const float* u     = (const float*)d_in[0];
    const float* gWih  = (const float*)d_in[1];
    const float* gWhh  = (const float*)d_in[2];
    const float* gbih  = (const float*)d_in[3];
    const float* gbhh  = (const float*)d_in[4];
    const float* Wage  = (const float*)d_in[5];
    const float* bage  = (const float*)d_in[6];
    const float* Wih   = (const float*)d_in[7];
    const float* Whh   = (const float*)d_in[8];
    const float* bih   = (const float*)d_in[9];
    const float* bhh   = (const float*)d_in[10];
    const float* tau   = (const float*)d_in[11];
    const float* Wout  = (const float*)d_in[12];
    const float* bout  = (const float*)d_in[13];
    float* out = (float*)d_out;

    char* ws = (char*)d_ws;
    unsigned short* whh_f = (unsigned short*)(ws);                 // 393216 B
    unsigned short* wih_f = (unsigned short*)(ws + 393216);        //  49152 B
    unsigned short* ht_f  = (unsigned short*)(ws + 442368);        // 131072 B
    float* A_ws  = (float*)(ws + 573440);                          // 67108864 B
    float* Bm_ws = (float*)(ws + 573440 + 67108864);               //  4194304 B
    float* C_ws  = (float*)(ws + 573440 + 67108864 + 4194304);     //   262144 B

    pack_weights_k<<<864, 256, 0, stream>>>(gWhh, gWih, whh_f, wih_f);

    const size_t GRU_LDS = 28672u * 4 + (4096u + 512u) * 2;        // 123904 B
    gru_k<<<16, 256, GRU_LDS, stream>>>(u, whh_f, wih_f, gbih, gbhh, ht_f);

    hyper_k<<<dim3(546, 16), 256, 0, stream>>>(ht_f, Wage, bage, Wih, Whh,
                                               bih, bhh, A_ws, Bm_ws, C_ws);

    const size_t DYN_LDS = (65536u + 4096u + 256u + 16u) * 4;      // 279616 B
    dyn_k<<<256, 256, DYN_LDS, stream>>>(u, A_ws, Bm_ws, C_ws, tau, Wout, bout, out);
}
